// GNNVAE_52905407152187
// MI455X (gfx1250) — matrix operand order repaired and checked by reading, not compiled, not run on hardware
//
#include <hip/hip_runtime.h>
#include <cstdint>
#include <cstddef>

typedef __attribute__((ext_vector_type(16))) _Float16 v16h;
typedef __attribute__((ext_vector_type(8)))  float    v8f;
typedef __attribute__((ext_vector_type(4)))  float    v4f;

#define TPB 256

// ---------------- degree / normalization ----------------

__global__ void k_fill1(float* __restrict__ p, int n) {
  int i = blockIdx.x * blockDim.x + threadIdx.x;
  if (i < n) p[i] = 1.0f;  // self-loop contributes 1 to every degree
}

__global__ void k_deg_edges(const int* __restrict__ dst, float* __restrict__ deg, int E) {
  int e = blockIdx.x * blockDim.x + threadIdx.x;
  if (e < E) unsafeAtomicAdd(&deg[dst[e]], 1.0f);
}

__global__ void k_rsqrt_inplace(float* __restrict__ p, int n) {
  int i = blockIdx.x * blockDim.x + threadIdx.x;
  if (i < n) p[i] = rsqrtf(p[i]);  // deg >= 1 always (self loops)
}

// ---------------- weight pack: f32 [K,N] -> f16 transposed [N,K] ----------------

__global__ void k_pack_bt(const float* __restrict__ W, _Float16* __restrict__ Bt,
                          int K, int N) {
  int t = blockIdx.x * blockDim.x + threadIdx.x;
  if (t < K * N) {
    int k = t / N, n = t - k * N;
    Bt[(size_t)n * K + k] = (_Float16)W[t];
  }
}

// ---------------- WMMA GEMM: C[M,N] = A[M,K](f32) * Bt[N,K](f16) (+bias)(+relu) ----
// One wave computes a 16 x (16*NT) strip: A loaded+converted ONCE per k-step and
// reused across NT v_wmma_f32_16x16x32_f16 issues (NT accumulators).
// A layout per lane (h=lane>>4, m=lane&15): a[i]   = A[m][k0+h*16+i]
// B layout per lane (n=lane&15):            b[i]   = B[k0+h*16+i][n]  (contiguous in Bt)
// C layout per lane:                        c[r] -> row m0+h*8+r, col n0+(lane&15)

template <int NT>
__global__ void k_wmma_gemm(const float* __restrict__ A, const _Float16* __restrict__ Bt,
                            const float* __restrict__ bias, float* __restrict__ C,
                            int M, int N, int K, int relu) {
  int wave = (blockIdx.x * blockDim.x + threadIdx.x) >> 5;
  int lane = threadIdx.x & 31;
  int ngroups = N / (16 * NT);
  int tm = wave / ngroups;
  int m0 = tm << 4;
  if (m0 >= M) return;                 // wave-uniform exit; EXEC stays all-ones for WMMA
  int n0 = (wave - tm * ngroups) * (16 * NT);
  int h = lane >> 4, l = lane & 15;

  const float* arow = A + (size_t)(m0 + l) * K + h * 16;
  const _Float16* brow[NT];
#pragma unroll
  for (int j = 0; j < NT; ++j)
    brow[j] = Bt + (size_t)(n0 + 16 * j + l) * K + h * 16;

  v8f c[NT];
#pragma unroll
  for (int j = 0; j < NT; ++j) c[j] = (v8f){};

  for (int k0 = 0; k0 < K; k0 += 32) {
    v4f a0 = *(const v4f*)(arow + k0);
    v4f a1 = *(const v4f*)(arow + k0 + 4);
    v4f a2 = *(const v4f*)(arow + k0 + 8);
    v4f a3 = *(const v4f*)(arow + k0 + 12);
    v16h a;
#pragma unroll
    for (int jj = 0; jj < 4; ++jj) {
      a[jj]      = (_Float16)a0[jj];
      a[4 + jj]  = (_Float16)a1[jj];
      a[8 + jj]  = (_Float16)a2[jj];
      a[12 + jj] = (_Float16)a3[jj];
    }
#pragma unroll
    for (int j = 0; j < NT; ++j) {
      v16h b = *(const v16h*)(brow[j] + k0);
      // (neg_a, A, neg_b, B, c_mod, C, reuse_a, reuse_b)
      c[j] = __builtin_amdgcn_wmma_f32_16x16x32_f16(false, a, false, b, (short)0,
                                                    c[j], false, false);
    }
  }

#pragma unroll
  for (int j = 0; j < NT; ++j) {
    int col = n0 + 16 * j + l;
    float bv = bias ? bias[col] : 0.0f;
    float* cp = C + (size_t)(m0 + h * 8) * N + col;
#pragma unroll
    for (int r = 0; r < 8; ++r) {
      float v = c[j][r] + bv;
      if (relu) v = fmaxf(v, 0.0f);
      cp[(size_t)r * N] = v;
    }
  }
}

// ---------------- edge scatter: acc[dst] += hp[src]*dinv[src] (4 feats/thread) -----

__global__ void k_scatter(const int* __restrict__ src, const int* __restrict__ dst,
                          const float* __restrict__ hp, const float* __restrict__ dinv,
                          float* __restrict__ acc, int E, int logF) {
  long long tid = (long long)blockIdx.x * blockDim.x + threadIdx.x;
  int shift = logF - 2;                      // threads per edge = F/4
  long long total = (long long)E << shift;
  if (tid >= total) return;
  int e = (int)(tid >> shift);
  int f = ((int)(tid & ((1LL << shift) - 1))) << 2;
  int F = 1 << logF;
  int s = src[e], d = dst[e];
  float w = dinv[s];
  v4f v = *(const v4f*)(hp + (size_t)s * F + f);
  float* ap = acc + (size_t)d * F + f;
  unsafeAtomicAdd(ap + 0, v[0] * w);
  unsafeAtomicAdd(ap + 1, v[1] * w);
  unsafeAtomicAdd(ap + 2, v[2] * w);
  unsafeAtomicAdd(ap + 3, v[3] * w);
}

// ---------------- finalize: out = dinv*(acc + hp*dinv) + bias (opt relu) ----------

__global__ void k_finalize(const float* __restrict__ hp, const float* __restrict__ acc,
                           const float* __restrict__ dinv, const float* __restrict__ bias,
                           float* __restrict__ out, int n, int F, int relu) {
  long long t = (long long)blockIdx.x * blockDim.x + threadIdx.x;
  if (t >= (long long)n * F) return;
  int i = (int)(t / F);
  int f = (int)(t - (long long)i * F);
  float di = dinv[i];
  float v = di * (acc[t] + hp[t] * di) + bias[f];
  if (relu) v = fmaxf(v, 0.0f);
  out[t] = v;
}

// ---------------- condition head: pred = z @ Wc[32,3] + bc ------------------------

__global__ void k_cond(const float* __restrict__ z, const float* __restrict__ Wc,
                       const float* __restrict__ bc, float* __restrict__ pred, int n) {
  long long t = (long long)blockIdx.x * blockDim.x + threadIdx.x;
  if (t >= 3LL * n) return;
  int i = (int)(t / 3);
  int j = (int)(t - 3LL * i);
  float s = bc[j];
  const float* zi = z + (size_t)i * 32;
#pragma unroll
  for (int k = 0; k < 32; ++k) s = fmaf(zi[k], Wc[k * 3 + j], s);
  pred[t] = s;
}

// ---------------- launch --------------------------------------------------------

static inline dim3 nblocks(long long threads) {
  return dim3((unsigned)((threads + TPB - 1) / TPB));
}

extern "C" void kernel_launch(void* const* d_in, const int* in_sizes, int n_in,
                              void* d_out, int out_size, void* d_ws, size_t ws_size,
                              hipStream_t stream) {
  const float* x  = (const float*)d_in[0];
  const int*   ei = (const int*)d_in[1];
  // d_in[2] = edge_attr: unused by reference
  const float* W1 = (const float*)d_in[3];
  const float* b1 = (const float*)d_in[4];
  const float* W2 = (const float*)d_in[5];
  const float* b2 = (const float*)d_in[6];
  const float* W3 = (const float*)d_in[7];
  const float* b3 = (const float*)d_in[8];
  const float* W4 = (const float*)d_in[9];
  const float* b4 = (const float*)d_in[10];
  const float* Wc = (const float*)d_in[11];
  const float* bc = (const float*)d_in[12];

  const int N = in_sizes[0] / 256;   // 50000 (multiple of 16)
  const int E = in_sizes[1] / 2;     // 800000
  const int* src = ei;
  const int* dst = ei + E;

  float* out_recon = (float*)d_out;                    // [N,256]
  float* out_z     = out_recon + (size_t)N * 256;      // [N,32]
  float* out_pred  = out_z + (size_t)N * 32;           // [N,3]
  (void)n_in; (void)out_size; (void)ws_size;

  // workspace carve-up (~129 MB total; hot set is L2-resident on MI455X)
  uint8_t* w = (uint8_t*)d_ws;
  auto carve = [&](size_t bytes) -> void* {
    void* p = (void*)w;
    w += (bytes + 255) & ~(size_t)255;
    return p;
  };
  float*    dinv  = (float*)carve((size_t)N * 4);
  _Float16* Bt1   = (_Float16*)carve((size_t)64 * 256 * 2);
  _Float16* Bt2   = (_Float16*)carve((size_t)32 * 64 * 2);
  _Float16* Bt3   = (_Float16*)carve((size_t)64 * 32 * 2);
  _Float16* Bt4   = (_Float16*)carve((size_t)256 * 64 * 2);
  float*    hp    = (float*)carve((size_t)N * 256 * 4);  // reused: [N,64] then [N,256]
  float*    acc   = (float*)carve((size_t)N * 256 * 4);  // reused accumulator
  float*    hbuf  = (float*)carve((size_t)N * 64 * 4);
  float*    hdbuf = (float*)carve((size_t)N * 64 * 4);

  // degrees -> dinv (in place)
  k_fill1<<<nblocks(N), TPB, 0, stream>>>(dinv, N);
  k_deg_edges<<<nblocks(E), TPB, 0, stream>>>(dst, dinv, E);
  k_rsqrt_inplace<<<nblocks(N), TPB, 0, stream>>>(dinv, N);

  // pack weights into f16 transposed [N,K] (WMMA B-operand friendly)
  k_pack_bt<<<nblocks(256 * 64), TPB, 0, stream>>>(W1, Bt1, 256, 64);
  k_pack_bt<<<nblocks(64 * 32),  TPB, 0, stream>>>(W2, Bt2, 64, 32);
  k_pack_bt<<<nblocks(32 * 64),  TPB, 0, stream>>>(W3, Bt3, 32, 64);
  k_pack_bt<<<nblocks(64 * 256), TPB, 0, stream>>>(W4, Bt4, 64, 256);

  // encoder GCN: hp = x @ W1   [N,64]  (NT=4: one wave covers all 64 cols)
  long long waves1 = (long long)(N / 16) * (64 / 64);
  k_wmma_gemm<4><<<nblocks(waves1 * 32), TPB, 0, stream>>>(x, Bt1, nullptr, hp, N, 64, 256, 0);
  hipMemsetAsync(acc, 0, (size_t)N * 64 * 4, stream);
  k_scatter<<<nblocks((long long)E << 4), TPB, 0, stream>>>(src, dst, hp, dinv, acc, E, 6);
  k_finalize<<<nblocks((long long)N * 64), TPB, 0, stream>>>(hp, acc, dinv, b1, hbuf, N, 64, 1);

  // z = h @ W2 + b2  -> d_out (also consumed below)  (NT=2: 32 cols per wave)
  long long waves2 = (long long)(N / 16);
  k_wmma_gemm<2><<<nblocks(waves2 * 32), TPB, 0, stream>>>(hbuf, Bt2, b2, out_z, N, 32, 64, 0);

  // hd = relu(z @ W3 + b3)   (NT=4)
  long long waves3 = (long long)(N / 16);
  k_wmma_gemm<4><<<nblocks(waves3 * 32), TPB, 0, stream>>>(out_z, Bt3, b3, hdbuf, N, 64, 32, 1);

  // decoder GCN: hp = hd @ W4   [N,256]  (NT=4: 4 n-groups of 64 cols)
  long long waves4 = (long long)(N / 16) * (256 / 64);
  k_wmma_gemm<4><<<nblocks(waves4 * 32), TPB, 0, stream>>>(hdbuf, Bt4, nullptr, hp, N, 256, 64, 0);
  hipMemsetAsync(acc, 0, (size_t)N * 256 * 4, stream);
  k_scatter<<<nblocks((long long)E << 6), TPB, 0, stream>>>(src, dst, hp, dinv, acc, E, 8);
  k_finalize<<<nblocks((long long)N * 256), TPB, 0, stream>>>(hp, acc, dinv, b4, out_recon, N, 256, 0);

  // condition head
  k_cond<<<nblocks((long long)N * 3), TPB, 0, stream>>>(out_z, Wc, bc, out_pred, N);
}
